// AlignerHead_83227876262532
// MI455X (gfx1250) — matrix operand order repaired and checked by reading, not compiled, not run on hardware
//
#include <hip/hip_runtime.h>
#include <hip/hip_bf16.h>

// ---------------------------------------------------------------------------
// Problem constants (from reference)
// ---------------------------------------------------------------------------
#define NFG    262144
#define NINST  1024
#define DM     256
#define NHEAD  4
#define DH     64
#define NLAYER 3

typedef __attribute__((ext_vector_type(16))) __bf16        v16bf;
typedef __attribute__((ext_vector_type(8)))  float         v8f;
typedef __attribute__((ext_vector_type(8)))  unsigned int  v8u;

__device__ __forceinline__ unsigned short f2bf(float x) {
    unsigned int u = __float_as_uint(x);
    unsigned int r = u + 0x7FFFu + ((u >> 16) & 1u);   // round-to-nearest-even
    return (unsigned short)(r >> 16);
}
__device__ __forceinline__ float bf_lo(unsigned int u) { return __uint_as_float(u << 16); }
__device__ __forceinline__ float bf_hi(unsigned int u) { return __uint_as_float(u & 0xFFFF0000u); }
__device__ __forceinline__ unsigned int f2key(float x) {  // order-preserving f32->u32
    unsigned int u = __float_as_uint(x);
    return (u & 0x80000000u) ? ~u : (u | 0x80000000u);
}
__device__ __forceinline__ float key2f(unsigned int k) {
    unsigned int u = (k & 0x80000000u) ? (k ^ 0x80000000u) : ~k;
    return __uint_as_float(u);
}

// ---------------------------------------------------------------------------
// Pack fp32 weight [Kreal x Nreal] row-major -> fragment-major bf16 B tiles.
// B fragment (32x16 bf16): lane = column (lane&15), half-wave = K half:
//   lane<16 covers K=ktile*32+0..15, lane>=16 covers K=ktile*32+16..31,
//   element e (0..15) -> K = base + e.  Out-of-range -> 0 (padding).
// ---------------------------------------------------------------------------
__global__ void pack_weights_kernel(const float* __restrict__ W, unsigned int* __restrict__ P,
                                    int Kreal, int Nreal, int Kpad, int Npad) {
    int numNT = Npad >> 4;
    int numKT = Kpad >> 5;
    int tid = blockIdx.x * blockDim.x + threadIdx.x;
    int total = numKT * numNT * 32;
    if (tid >= total) return;
    int lane = tid & 31;
    int tile = tid >> 5;
    int nt = tile % numNT, kt = tile / numNT;
    int n  = (nt << 4) + (lane & 15);
    int kb = (kt << 5) + ((lane >> 4) << 4);
    unsigned int o[8];
#pragma unroll
    for (int j = 0; j < 8; ++j) {
        int k0 = kb + (j << 1);
        unsigned short lo = 0, hi = 0;
        if (n < Nreal) {
            if (k0     < Kreal) lo = f2bf(W[(size_t)k0 * Nreal + n]);
            if (k0 + 1 < Kreal) hi = f2bf(W[(size_t)(k0 + 1) * Nreal + n]);
        }
        o[j] = (unsigned int)lo | ((unsigned int)hi << 16);
    }
    unsigned int* dst = P + ((size_t)tile * 32 + lane) * 8;
#pragma unroll
    for (int j = 0; j < 8; ++j) dst[j] = o[j];
}

// ---------------------------------------------------------------------------
// Generic bf16 WMMA GEMM:  out[M x Nreal] = A[M x Kpad](bf16) @ Bpacked (+bias)(+residual)
// Block = 256 threads (8 waves). Block tile 128 x (NTILES*16). Wave tile 16 x NTILES*16.
// NTILES is compile-time so all accumulators stay in VGPRs (8 VGPRs per tile).
// Optional epilogues: f32 store, bf16 store, per-column sum / sumsq atomics.
// M multiple of 128; Kpad multiple of 32; Npad == gridDim.y * NTILES * 16.
// ---------------------------------------------------------------------------
template <int NTILES>
__global__ __launch_bounds__(256)
void gemm_bf16_wmma(const unsigned short* __restrict__ A,
                    const unsigned int*  __restrict__ Pw,
                    int M, int Kpad, int Npad, int Nreal,
                    const float* __restrict__ bias,
                    const float* __restrict__ residual,
                    float* __restrict__ outF32,
                    unsigned short* __restrict__ outBF,
                    float* __restrict__ colsum,
                    float* __restrict__ colsumsq) {
    __shared__ unsigned short As[128][40];   // stride 40 shorts = 80B: 16B aligned, conflict-free

    const int lane = threadIdx.x & 31;
    const int wave = threadIdx.x >> 5;
    const int rowBase = blockIdx.x * 128;
    const int numNT = Npad >> 4;
    const int colTileBase = blockIdx.y * NTILES;
    const int numKT = Kpad >> 5;
    const int mloc = lane & 15;
    const int hf   = lane >> 4;

    v8f acc[NTILES];
#pragma unroll
    for (int t = 0; t < NTILES; ++t)
#pragma unroll
        for (int r = 0; r < 8; ++r) acc[t][r] = 0.0f;

    for (int kt = 0; kt < numKT; ++kt) {
        __syncthreads();
        {   // cooperative 128x32 bf16 A tile load (fully coalesced uint4)
            int r  = threadIdx.x >> 1;
            int h2 = threadIdx.x & 1;
            const uint4* src = (const uint4*)(A + (size_t)(rowBase + r) * Kpad + (kt << 5) + (h2 << 4));
            uint4 x0 = src[0], x1 = src[1];
            if (kt + 1 < numKT)
                __builtin_prefetch(A + (size_t)(rowBase + r) * Kpad + ((kt + 1) << 5), 0, 1);
            uint4* dst = (uint4*)(&As[r][h2 << 4]);
            dst[0] = x0; dst[1] = x1;
        }
        __syncthreads();

        // A fragment (16x32): lane row = lane&15, K = 16*(lane>>4) + e (contiguous, matches packed B)
        // (two contiguous 16B chunks per lane -> compiler emits 2x ds_load_b128)
        v8u au;
        const unsigned short* arow = &As[(wave << 4) + mloc][0];
#pragma unroll
        for (int j = 0; j < 8; ++j) {
            int k0 = (hf << 4) + (j << 1);
            au[j] = *(const unsigned int*)(arow + k0);
        }
        v16bf av = __builtin_bit_cast(v16bf, au);

#pragma unroll
        for (int t = 0; t < NTILES; ++t) {
            size_t tile = (size_t)kt * numNT + (colTileBase + t);
            v8u bu = *(const v8u*)(Pw + (tile * 32 + lane) * 8);   // 32B, aligned
            v16bf bv = __builtin_bit_cast(v16bf, bu);
            acc[t] = __builtin_amdgcn_wmma_f32_16x16x32_bf16(
                false, av, false, bv, (short)0, acc[t], false, false);
        }
    }

    // Epilogue: C layout: lane col n = lane&15; vgpr r -> row r + 8*(lane>>4)
#pragma unroll
    for (int t = 0; t < NTILES; ++t) {
        int n = ((colTileBase + t) << 4) + mloc;
        bool nOK = n < Nreal;
        float bia = (bias != nullptr && nOK) ? bias[n] : 0.0f;
        float cs = 0.0f, cq = 0.0f;
#pragma unroll
        for (int r = 0; r < 8; ++r) {
            int m = rowBase + (wave << 4) + r + (hf << 3);
            float val = acc[t][r] + bia;
            if (nOK) {
                size_t o = (size_t)m * Nreal + n;
                if (residual) val += residual[o];
                if (outF32) outF32[o] = val;
                if (outBF)  outBF[o] = f2bf(val);
            }
            cs += val; cq += val * val;
        }
        if (colsum && nOK) {
            atomicAdd(&colsum[n], cs);
            atomicAdd(&colsumsq[n], cq);
        }
    }
}

// ---------------------------------------------------------------------------
// Build bf16 GEMM input for fg MLP: row = [fg_feat(256) | pos_embed(27) | 0 x5]
// One wave per fg point.
// ---------------------------------------------------------------------------
__global__ void build_A_fg_kernel(const float* __restrict__ boxes, const float* __restrict__ fg,
                                  const float* __restrict__ fgfeat, const int* __restrict__ idx,
                                  unsigned short* __restrict__ Aout) {
    int gid = blockIdx.x * blockDim.x + threadIdx.x;
    int wid = gid >> 5, lane = gid & 31;
    if (wid >= NFG) return;
    unsigned short* row = Aout + (size_t)wid * 288;

    // fg_feat -> bf16 (coalesced float4 loads, uint4 store)
    const float* f = fgfeat + (size_t)wid * 256 + lane * 8;
    float4 f0 = *(const float4*)f;
    float4 f1 = *(const float4*)(f + 4);
    uint4 pk;
    pk.x = (unsigned)f2bf(f0.x) | ((unsigned)f2bf(f0.y) << 16);
    pk.y = (unsigned)f2bf(f0.z) | ((unsigned)f2bf(f0.w) << 16);
    pk.z = (unsigned)f2bf(f1.x) | ((unsigned)f2bf(f1.y) << 16);
    pk.w = (unsigned)f2bf(f1.z) | ((unsigned)f2bf(f1.w) << 16);
    *(uint4*)(row + lane * 8) = pk;

    // pos embed
    int i = idx[wid];
    const float* bx = boxes + (size_t)i * 7;
    float cx = bx[0], cy = bx[1], cz = bx[2];
    float dx = bx[3], dy = bx[4], dz = bx[5], yaw = bx[6];
    float sn, cs; sincosf(yaw, &sn, &cs);
    const float* p = fg + (size_t)wid * 5;
    float px = p[1], py = p[2], pz = p[3];
    float fib0 = cs * px - sn * py + cx;
    float fib1 = sn * px + cs * py + cy;
    float fib2 = pz + cz;
    if (lane < 27) {
        float val;
        if (lane < 3) {
            val = (lane == 0) ? fib0 : ((lane == 1) ? fib1 : fib2);
        } else {
            int j = lane - 3, cc = j / 3, a = j - cc * 3;
            float sx = (cc < 4) ? 1.f : -1.f;
            int c4 = cc & 3;
            float sy = (c4 == 1 || c4 == 2) ? 1.f : -1.f;
            float sz = (cc & 2) ? -1.f : 1.f;
            float s3 = (a == 0) ? sx : ((a == 1) ? sy : sz);
            float fa = (a == 0) ? fib0 : ((a == 1) ? fib1 : fib2);
            float da = (a == 0) ? dx : ((a == 1) ? dy : dz);
            val = fa - 0.5f * s3 * da;
        }
        row[256 + lane] = f2bf(val);
    } else {
        row[256 + lane] = 0;   // pad K 283..287
    }
}

// ---------------------------------------------------------------------------
// Small elementwise kernels
// ---------------------------------------------------------------------------
__global__ void f32_to_bf16_kernel(const float* __restrict__ in, unsigned short* __restrict__ out, int n) {
    int t = blockIdx.x * blockDim.x + threadIdx.x;
    if (t < n) out[t] = f2bf(in[t]);
}

__global__ void bn_finalize_kernel(const float* __restrict__ colsum, const float* __restrict__ colsq,
                                   const float* __restrict__ g, const float* __restrict__ b,
                                   float invM, float* __restrict__ scale, float* __restrict__ shift, int N) {
    int n = blockIdx.x * blockDim.x + threadIdx.x;
    if (n >= N) return;
    float mean = colsum[n] * invM;
    float var  = colsq[n] * invM - mean * mean;
    float s = g[n] * rsqrtf(var + 0.001f);
    scale[n] = s;
    shift[n] = b[n] - mean * s;
}

__global__ void bn_relu_bf16_kernel(const float* __restrict__ h, const float* __restrict__ scale,
                                    const float* __restrict__ shift, unsigned short* __restrict__ out,
                                    int total, int N) {
    int t = blockIdx.x * blockDim.x + threadIdx.x;
    if (t >= total) return;
    int n = t % N;
    float v = h[t] * scale[n] + shift[n];
    out[t] = f2bf(fmaxf(v, 0.0f));
}

// ---------------------------------------------------------------------------
// Attention. k and v are stored bf16 (GEMM epilogue) to halve HBM traffic.
// Wave per fg point: lane owns 8 channels (16B = one uint4 of 8 bf16).
// ---------------------------------------------------------------------------
__global__ void logits_kernel(const float* __restrict__ q, const unsigned short* __restrict__ kbf,
                              const int* __restrict__ idx, float* __restrict__ logits,
                              unsigned int* __restrict__ segmax) {
    int gid = blockIdx.x * blockDim.x + threadIdx.x;
    int wid = gid >> 5, lane = gid & 31;
    if (wid >= NFG) return;
    int i = idx[wid];
    uint4 kv = *(const uint4*)(kbf + (size_t)wid * 256 + lane * 8);   // 8 bf16
    const float* qr = q + (size_t)i * 256 + lane * 8;
    float4 b0 = *(const float4*)qr, b1 = *(const float4*)(qr + 4);
    float s = bf_lo(kv.x) * b0.x + bf_hi(kv.x) * b0.y
            + bf_lo(kv.y) * b0.z + bf_hi(kv.y) * b0.w
            + bf_lo(kv.z) * b1.x + bf_hi(kv.z) * b1.y
            + bf_lo(kv.w) * b1.z + bf_hi(kv.w) * b1.w;
    s += __shfl_xor(s, 1);
    s += __shfl_xor(s, 2);
    s += __shfl_xor(s, 4);                 // 8-lane group = one head (DH=64)
    if ((lane & 7) == 0) {
        int h = lane >> 3;
        float lg = s * 0.125f;             // 1/sqrt(64)
        logits[(size_t)wid * NHEAD + h] = lg;
        atomicMax(&segmax[i * NHEAD + h], f2key(lg));
    }
}

// exp, segment-sum, and weighted-v segment accumulation (wave per point)
__global__ void expsum_ev_kernel(const float* __restrict__ logits, const unsigned int* __restrict__ segmax,
                                 const unsigned short* __restrict__ vbf, const int* __restrict__ idx,
                                 float* __restrict__ evec, float* __restrict__ segsum,
                                 float* __restrict__ ctx) {
    int gid = blockIdx.x * blockDim.x + threadIdx.x;
    int wid = gid >> 5, lane = gid & 31;
    if (wid >= NFG) return;
    int i = idx[wid];
    int h = lane >> 3;
    float m  = key2f(segmax[i * NHEAD + h]);
    float lg = logits[(size_t)wid * NHEAD + h];
    float e  = expf(lg - m);
    if ((lane & 7) == 0) {
        evec[(size_t)wid * NHEAD + h] = e;
        atomicAdd(&segsum[i * NHEAD + h], e);
    }
    uint4 vv = *(const uint4*)(vbf + (size_t)wid * 256 + lane * 8);   // 8 bf16
    float* cr = ctx + (size_t)i * 256 + lane * 8;
    atomicAdd(&cr[0], e * bf_lo(vv.x)); atomicAdd(&cr[1], e * bf_hi(vv.x));
    atomicAdd(&cr[2], e * bf_lo(vv.y)); atomicAdd(&cr[3], e * bf_hi(vv.y));
    atomicAdd(&cr[4], e * bf_lo(vv.z)); atomicAdd(&cr[5], e * bf_hi(vv.z));
    atomicAdd(&cr[6], e * bf_lo(vv.w)); atomicAdd(&cr[7], e * bf_hi(vv.w));
}

__global__ void ctx_finalize_kernel(const float* __restrict__ ctx, const float* __restrict__ segsum,
                                    unsigned short* __restrict__ ctx_bf) {
    int t = blockIdx.x * blockDim.x + threadIdx.x;
    if (t >= NINST * 256) return;
    int i = t >> 8, c = t & 255, h = c >> 6;
    float s = fmaxf(segsum[i * NHEAD + h], 1e-12f);
    ctx_bf[t] = f2bf(ctx[t] / s);
}

__global__ void attn_out_kernel(const float* __restrict__ evec, const float* __restrict__ segsum,
                                const int* __restrict__ idx, float* __restrict__ out) {
    int t = blockIdx.x * blockDim.x + threadIdx.x;
    if (t >= NFG * NHEAD) return;
    int n = t >> 2, h = t & 3;
    int i = idx[n];
    out[t] = evec[t] / fmaxf(segsum[i * NHEAD + h], 1e-12f);
}

// ---------------------------------------------------------------------------
// Orchestration
// ---------------------------------------------------------------------------
extern "C" void kernel_launch(void* const* d_in, const int* in_sizes, int n_in,
                              void* d_out, int out_size, void* d_ws, size_t ws_size,
                              hipStream_t stream) {
    const float* pred_boxes = (const float*)d_in[0];
    const float* fg         = (const float*)d_in[1];
    const float* fg_feat    = (const float*)d_in[2];
    const float* shape_enc  = (const float*)d_in[3];
    const float* enc_fg_W1  = (const float*)d_in[4];
    const float* enc_fg_g   = (const float*)d_in[5];
    const float* enc_fg_b   = (const float*)d_in[6];
    const float* enc_fg_W2  = (const float*)d_in[7];
    const float* enc_fg_b2  = (const float*)d_in[8];
    const float* enc_in_W1  = (const float*)d_in[9];
    const float* enc_in_g   = (const float*)d_in[10];
    const float* enc_in_b   = (const float*)d_in[11];
    const float* enc_in_W2  = (const float*)d_in[12];
    const float* enc_in_b2  = (const float*)d_in[13];
    const float* dec_W1     = (const float*)d_in[14];
    const float* dec_g      = (const float*)d_in[15];
    const float* dec_b      = (const float*)d_in[16];
    const float* dec_W2     = (const float*)d_in[17];
    const float* dec_b2     = (const float*)d_in[18];
    const float* Wq         = (const float*)d_in[19];
    const float* Wk         = (const float*)d_in[20];
    const float* Wv         = (const float*)d_in[21];
    const float* Wo         = (const float*)d_in[22];
    const float* bq         = (const float*)d_in[23];
    const float* bk         = (const float*)d_in[24];
    const float* bv         = (const float*)d_in[25];
    const float* bo         = (const float*)d_in[26];
    const int*   inst_idx   = (const int*)d_in[27];
    float* out = (float*)d_out;             // [1024*8 refine][262144*4 attn]

    // ---- workspace carve ----
    char* base = (char*)d_ws;
    size_t off = 0;
    auto carve = [&](size_t bytes) -> char* {
        char* p = base + off;
        off += (bytes + 255) & ~(size_t)255;
        return p;
    };
    unsigned short* A_big  = (unsigned short*)carve((size_t)NFG * 288 * 2); // A_fg, later A2
    float*          h_big  = (float*)carve((size_t)NFG * 256 * 4);          // h fp32, later k bf16
    unsigned short* fgf_bf = (unsigned short*)carve((size_t)NFG * 256 * 2);
    unsigned short* v_bf   = (unsigned short*)carve((size_t)NFG * 256 * 2);
    float*          logitb = (float*)carve((size_t)NFG * NHEAD * 4);
    float*          evec   = (float*)carve((size_t)NFG * NHEAD * 4);
    float*          inst   = (float*)carve((size_t)NINST * 256 * 4);
    unsigned short* instbf = (unsigned short*)carve((size_t)NINST * 256 * 2);
    float*          qbuf   = (float*)carve((size_t)NINST * 256 * 4);
    float*          ctx    = (float*)carve((size_t)NINST * 256 * 4);
    unsigned short* ctxbf  = (unsigned short*)carve((size_t)NINST * 256 * 2);
    float*          h_sm   = (float*)carve((size_t)NINST * 256 * 4);        // inst-MLP h, dec h
    unsigned short* a_sm   = (unsigned short*)carve((size_t)NINST * 256 * 2);
    float*          colsum = (float*)carve(256 * 4);
    float*          colsq  = (float*)carve(256 * 4);
    float*          scaleb = (float*)carve(256 * 4);
    float*          shiftb = (float*)carve(256 * 4);
    unsigned int*   segmax = (unsigned int*)carve(NINST * NHEAD * 4);
    float*          segsum = (float*)carve(NINST * NHEAD * 4);
    unsigned int*   P_fg1  = (unsigned int*)carve((size_t)288 * 256 * 2);
    unsigned int*   P_fg2  = (unsigned int*)carve((size_t)256 * 256 * 2);
    unsigned int*   P_in1  = (unsigned int*)carve((size_t)256 * 256 * 2);
    unsigned int*   P_in2  = (unsigned int*)carve((size_t)256 * 256 * 2);
    unsigned int*   P_dec1 = (unsigned int*)carve((size_t)256 * 128 * 2);
    unsigned int*   P_dec2 = (unsigned int*)carve((size_t)128 * 16 * 2);
    unsigned int *P_q[NLAYER], *P_k[NLAYER], *P_v[NLAYER], *P_o[NLAYER];
    for (int l = 0; l < NLAYER; ++l) {
        P_q[l] = (unsigned int*)carve((size_t)256 * 256 * 2);
        P_k[l] = (unsigned int*)carve((size_t)256 * 256 * 2);
        P_v[l] = (unsigned int*)carve((size_t)256 * 256 * 2);
        P_o[l] = (unsigned int*)carve((size_t)256 * 256 * 2);
    }
    unsigned short* k_bf = (unsigned short*)h_big;   // reuse h region after BN pass

    auto packW = [&](const float* W, unsigned int* P, int Kr, int Nr, int Kp, int Np) {
        int total = (Kp / 32) * (Np / 16) * 32;
        pack_weights_kernel<<<(total + 255) / 256, 256, 0, stream>>>(W, P, Kr, Nr, Kp, Np);
    };
    // Npad in {256,128} -> NTILES=8; Npad==16 -> NTILES=1. Compile-time tiles keep acc in VGPRs.
    auto gemm = [&](const unsigned short* A, const unsigned int* P, int M, int Kp, int Np, int Nr,
                    const float* bias, const float* res, float* oF, unsigned short* oB,
                    float* cs, float* cq) {
        if (Np >= 128) {
            dim3 g(M / 128, Np / 128);
            gemm_bf16_wmma<8><<<g, 256, 0, stream>>>(A, P, M, Kp, Np, Nr, bias, res, oF, oB, cs, cq);
        } else {
            dim3 g(M / 128, Np / 16);
            gemm_bf16_wmma<1><<<g, 256, 0, stream>>>(A, P, M, Kp, Np, Nr, bias, res, oF, oB, cs, cq);
        }
    };

    // ---- pack all weights (bf16 fragment-major) ----
    packW(enc_fg_W1, P_fg1, 283, 256, 288, 256);
    packW(enc_fg_W2, P_fg2, 256, 256, 256, 256);
    packW(enc_in_W1, P_in1, 256, 256, 256, 256);
    packW(enc_in_W2, P_in2, 256, 256, 256, 256);
    packW(dec_W1,    P_dec1, 256, 128, 256, 128);
    packW(dec_W2,    P_dec2, 128, 8, 128, 16);
    for (int l = 0; l < NLAYER; ++l) {
        packW(Wq + (size_t)l * 256 * 256, P_q[l], 256, 256, 256, 256);
        packW(Wk + (size_t)l * 256 * 256, P_k[l], 256, 256, 256, 256);
        packW(Wv + (size_t)l * 256 * 256, P_v[l], 256, 256, 256, 256);
        packW(Wo + (size_t)l * 256 * 256, P_o[l], 256, 256, 256, 256);
    }

    // ---- fg branch: [fg_feat|pe] -> MLP with batch-dim BN ----
    build_A_fg_kernel<<<(NFG * 32) / 256, 256, 0, stream>>>(pred_boxes, fg, fg_feat, inst_idx, A_big);
    hipMemsetAsync(colsum, 0, 256 * 4, stream);
    hipMemsetAsync(colsq,  0, 256 * 4, stream);
    gemm(A_big, P_fg1, NFG, 288, 256, 256, nullptr, nullptr, h_big, nullptr, colsum, colsq);
    bn_finalize_kernel<<<1, 256, 0, stream>>>(colsum, colsq, enc_fg_g, enc_fg_b,
                                              1.0f / (float)NFG, scaleb, shiftb, 256);
    bn_relu_bf16_kernel<<<(NFG * 256) / 256, 256, 0, stream>>>(h_big, scaleb, shiftb, A_big, NFG * 256, 256);
    gemm(A_big, P_fg2, NFG, 256, 256, 256, enc_fg_b2, nullptr, nullptr, fgf_bf, nullptr, nullptr);

    // ---- inst branch ----
    f32_to_bf16_kernel<<<(NINST * 256) / 256, 256, 0, stream>>>(shape_enc, a_sm, NINST * 256);
    hipMemsetAsync(colsum, 0, 256 * 4, stream);
    hipMemsetAsync(colsq,  0, 256 * 4, stream);
    gemm(a_sm, P_in1, NINST, 256, 256, 256, nullptr, nullptr, h_sm, nullptr, colsum, colsq);
    bn_finalize_kernel<<<1, 256, 0, stream>>>(colsum, colsq, enc_in_g, enc_in_b,
                                              1.0f / (float)NINST, scaleb, shiftb, 256);
    bn_relu_bf16_kernel<<<(NINST * 256) / 256, 256, 0, stream>>>(h_sm, scaleb, shiftb, a_sm, NINST * 256, 256);
    gemm(a_sm, P_in2, NINST, 256, 256, 256, enc_in_b2, nullptr, inst, nullptr, nullptr, nullptr);

    // ---- grouped MHA layers (k/v stored bf16: halves HBM traffic of the big streams) ----
    for (int l = 0; l < NLAYER; ++l) {
        f32_to_bf16_kernel<<<(NINST * 256) / 256, 256, 0, stream>>>(inst, instbf, NINST * 256);
        gemm(instbf, P_q[l], NINST, 256, 256, 256, bq + l * 256, nullptr, qbuf, nullptr, nullptr, nullptr);
        gemm(fgf_bf, P_k[l], NFG, 256, 256, 256, bk + l * 256, nullptr, nullptr, k_bf, nullptr, nullptr);
        gemm(fgf_bf, P_v[l], NFG, 256, 256, 256, bv + l * 256, nullptr, nullptr, v_bf, nullptr, nullptr);

        hipMemsetAsync(segmax, 0, NINST * NHEAD * 4, stream);   // 0 == key(-NaN): safe minimum
        hipMemsetAsync(segsum, 0, NINST * NHEAD * 4, stream);
        hipMemsetAsync(ctx,    0, (size_t)NINST * 256 * 4, stream);

        logits_kernel<<<(NFG * 32) / 256, 256, 0, stream>>>(qbuf, k_bf, inst_idx, logitb, segmax);
        expsum_ev_kernel<<<(NFG * 32) / 256, 256, 0, stream>>>(logitb, segmax, v_bf, inst_idx,
                                                               evec, segsum, ctx);
        ctx_finalize_kernel<<<(NINST * 256) / 256, 256, 0, stream>>>(ctx, segsum, ctxbf);
        // inst = inst + ctx @ Wo + bo  (residual epilogue, in-place)
        gemm(ctxbf, P_o[l], NINST, 256, 256, 256, bo + l * 256, inst, inst, nullptr, nullptr, nullptr);
    }

    // ---- attn output (last layer) ----
    attn_out_kernel<<<(NFG * NHEAD) / 256, 256, 0, stream>>>(evec, segsum, inst_idx, out + NINST * 8);

    // ---- decoder MLP -> refine ----
    f32_to_bf16_kernel<<<(NINST * 256) / 256, 256, 0, stream>>>(inst, instbf, NINST * 256);
    hipMemsetAsync(colsum, 0, 256 * 4, stream);
    hipMemsetAsync(colsq,  0, 256 * 4, stream);
    gemm(instbf, P_dec1, NINST, 256, 128, 128, nullptr, nullptr, h_sm, nullptr, colsum, colsq);
    bn_finalize_kernel<<<1, 256, 0, stream>>>(colsum, colsq, dec_g, dec_b,
                                              1.0f / (float)NINST, scaleb, shiftb, 128);
    bn_relu_bf16_kernel<<<(NINST * 128) / 256, 256, 0, stream>>>(h_sm, scaleb, shiftb, a_sm, NINST * 128, 128);
    gemm(a_sm, P_dec2, NINST, 128, 16, 8, dec_b2, nullptr, out, nullptr, nullptr, nullptr);
}
